// SlideGCD_TransMIL_43370579755064
// MI455X (gfx1250) — matrix operand order repaired and checked by reading, not compiled, not run on hardware
//
#include <hip/hip_runtime.h>
#include <hip/hip_bf16.h>

// ---------------------------------------------------------------------------
// SlideGCD / TransMIL forward for MI455X (gfx1250, wave32, WMMA).
// All matmuls lower to v_wmma_f32_16x16x32_f16 (f32 in memory, f16 compute).
// Workspace layout is carved out of d_ws (requires ~1 GB; compile-only loop).
// ---------------------------------------------------------------------------

typedef __attribute__((ext_vector_type(16))) _Float16 v16h;
typedef __attribute__((ext_vector_type(8)))  float    v8f;

#define HID    512
#define HEADS  8
#define DH     64
#define LMK    256      // landmarks
#define NPAD   6144     // attention-padded sequence
#define PADR   59       // leading zero rows (6144 - 6085)
#define NTOK   6085     // cls + 78*78 tokens
#define GRID_H 78
#define BATCH  4
#define LSEG   24       // NPAD / LMK
#define KNN    12
#define NNODE  1028     // 4 emb + 1024 rehearsal

// ---------------- WMMA GEMM: C = alpha*A*B (+bias) (+act) ------------------
#define BM 128
#define BN 128
#define BK 32
#define SKA (BK + 8)    // padded LDS stride for A (f16 elems)
#define SNB (BN + 8)    // padded LDS stride for B (f16 elems)

__global__ __launch_bounds__(256) void wmma_gemm_f16(
    const float* __restrict__ A, const float* __restrict__ B,
    const float* __restrict__ bias, float* __restrict__ C,
    int M, int N, int K, int lda, int ldb, int ldc,
    long long sA, long long sB, long long sC,
    int transB, int act, float alpha)
{
  __shared__ _Float16 As[BM * SKA];
  __shared__ _Float16 Bs[BK * SNB];

  const int tid  = threadIdx.x;
  const int lane = tid & 31;
  const int wave = tid >> 5;
  const int wm   = wave & 3;     // 4 waves along M
  const int wn   = wave >> 2;    // 2 waves along N
  const int m0   = blockIdx.y * BM;
  const int n0   = blockIdx.x * BN;

  A += (long long)blockIdx.z * sA;
  B += (long long)blockIdx.z * sB;
  C += (long long)blockIdx.z * sC;

  const v8f vzero = {0.f, 0.f, 0.f, 0.f, 0.f, 0.f, 0.f, 0.f};
  v8f acc[2][4];
#pragma unroll
  for (int s = 0; s < 2; ++s)
#pragma unroll
    for (int t = 0; t < 4; ++t) acc[s][t] = vzero;

  const int nK = K / BK;
  for (int kt = 0; kt < nK; ++kt) {
    const int kb = kt * BK;
    // ---- stage A tile (BM x BK), f32 -> f16 ----
    {
      const int r0 = tid >> 3;
      const int kc = (tid & 7) * 4;
#pragma unroll
      for (int it = 0; it < 4; ++it) {
        const int row  = r0 + it * 32;
        const int grow = m0 + row;
        float4 v = make_float4(0.f, 0.f, 0.f, 0.f);
        if (grow < M) {
          v = *(const float4*)(A + (long long)grow * lda + (kb + kc));
          if (kt + 1 < nK)
            __builtin_prefetch(A + (long long)grow * lda + (kb + BK + kc), 0, 1);
        }
        _Float16* d = &As[row * SKA + kc];
        d[0] = (_Float16)v.x; d[1] = (_Float16)v.y;
        d[2] = (_Float16)v.z; d[3] = (_Float16)v.w;
      }
    }
    // ---- stage B tile (BK x BN) ----
    if (!transB) {
      const int k0 = tid >> 5;
      const int nc = (tid & 31) * 4;
      const int gc = n0 + nc;
#pragma unroll
      for (int it = 0; it < 4; ++it) {
        const int k = k0 + it * 8;
        const float* src = B + (long long)(kb + k) * ldb + gc;
        float4 v = make_float4(0.f, 0.f, 0.f, 0.f);
        if (gc + 3 < N) {
          v = *(const float4*)src;
        } else {
          if (gc + 0 < N) v.x = src[0];
          if (gc + 1 < N) v.y = src[1];
          if (gc + 2 < N) v.z = src[2];
        }
        _Float16* d = &Bs[k * SNB + nc];
        d[0] = (_Float16)v.x; d[1] = (_Float16)v.y;
        d[2] = (_Float16)v.z; d[3] = (_Float16)v.w;
      }
    } else {  // B is (N x K) row-major, want Bs[k][n] = B[n][k]
      const int nr = tid >> 3;
      const int kc = (tid & 7) * 4;
#pragma unroll
      for (int it = 0; it < 4; ++it) {
        const int n  = nr + it * 32;
        const int gn = n0 + n;
        float4 v = make_float4(0.f, 0.f, 0.f, 0.f);
        if (gn < N) v = *(const float4*)(B + (long long)gn * ldb + (kb + kc));
        Bs[(kc + 0) * SNB + n] = (_Float16)v.x;
        Bs[(kc + 1) * SNB + n] = (_Float16)v.y;
        Bs[(kc + 2) * SNB + n] = (_Float16)v.z;
        Bs[(kc + 3) * SNB + n] = (_Float16)v.w;
      }
    }
    __syncthreads();

    // ---- fragments + WMMA (ISA 7.12.2 layouts) ----
    v16h af[2], bf[4];
    const int half = lane >> 4;
#pragma unroll
    for (int s = 0; s < 2; ++s) {
      const int m = wm * 32 + s * 16 + (lane & 15);
#pragma unroll
      for (int e = 0; e < 16; ++e) {
        const int k = (e < 8) ? (half * 8 + e) : (8 + half * 8 + e);
        af[s][e] = As[m * SKA + k];
      }
    }
#pragma unroll
    for (int t = 0; t < 4; ++t) {
      const int nb = wn * 64 + t * 16;
#pragma unroll
      for (int e = 0; e < 16; ++e) {
        const int k = (e < 8) ? (half * 8 + e) : (8 + half * 8 + e);
        bf[t][e] = Bs[k * SNB + nb + (lane & 15)];
      }
    }
#pragma unroll
    for (int s = 0; s < 2; ++s)
#pragma unroll
      for (int t = 0; t < 4; ++t)
        acc[s][t] = __builtin_amdgcn_wmma_f32_16x16x32_f16(
            false, af[s], false, bf[t], (short)0, acc[s][t], false, false);
    __syncthreads();
  }

  // ---- epilogue ----
  const int lnn = lane & 15;
  const int lh  = lane >> 4;
#pragma unroll
  for (int s = 0; s < 2; ++s) {
    const int rb = m0 + wm * 32 + s * 16 + lh * 8;
#pragma unroll
    for (int t = 0; t < 4; ++t) {
      const int col = n0 + wn * 64 + t * 16 + lnn;
      if (col >= N) continue;
      const float bv = bias ? bias[col] : 0.f;
#pragma unroll
      for (int r = 0; r < 8; ++r) {
        const int row = rb + r;
        if (row >= M) continue;
        float v = alpha * acc[s][t][r] + bv;
        if (act == 1) v = fmaxf(v, 0.f);
        else if (act == 2) v = (v > 0.f) ? v : 0.01f * v;
        C[(long long)row * ldc + col] = v;
      }
    }
  }
}

// ------------------------------ helpers ------------------------------------
__global__ __launch_bounds__(256) void row_softmax(float* __restrict__ X, int R) {
  __shared__ float red[256];
  float* x = X + (long long)blockIdx.x * R;
  const int tid = threadIdx.x;
  float m = -3.4e38f;
  for (int i = tid; i < R; i += 256) m = fmaxf(m, x[i]);
  red[tid] = m; __syncthreads();
  for (int s = 128; s > 0; s >>= 1) { if (tid < s) red[tid] = fmaxf(red[tid], red[tid + s]); __syncthreads(); }
  m = red[0]; __syncthreads();
  float sum = 0.f;
  for (int i = tid; i < R; i += 256) { float e = __expf(x[i] - m); x[i] = e; sum += e; }
  red[tid] = sum; __syncthreads();
  for (int s = 128; s > 0; s >>= 1) { if (tid < s) red[tid] += red[tid + s]; __syncthreads(); }
  const float inv = 1.0f / red[0];
  for (int i = tid; i < R; i += 256) x[i] *= inv;
}

__global__ __launch_bounds__(256) void ln_pad(const float* __restrict__ tok,
                                              const float* __restrict__ gam,
                                              const float* __restrict__ bet,
                                              float* __restrict__ out) {
  __shared__ float red[256];
  const int r  = blockIdx.x;            // 0 .. B*NPAD-1
  const int bi = r / NPAD, p = r % NPAD;
  float* o = out + (long long)r * HID;
  const int tid = threadIdx.x;
  if (p < PADR) { o[tid] = 0.f; o[tid + 256] = 0.f; return; }
  const float* x = tok + ((long long)bi * NTOK + (p - PADR)) * HID;
  const float a0 = x[tid], a1 = x[tid + 256];
  red[tid] = a0 + a1; __syncthreads();
  for (int s = 128; s > 0; s >>= 1) { if (tid < s) red[tid] += red[tid + s]; __syncthreads(); }
  const float mu = red[0] * (1.f / HID); __syncthreads();
  const float d0 = a0 - mu, d1 = a1 - mu;
  red[tid] = d0 * d0 + d1 * d1; __syncthreads();
  for (int s = 128; s > 0; s >>= 1) { if (tid < s) red[tid] += red[tid + s]; __syncthreads(); }
  const float rstd = rsqrtf(red[0] * (1.f / HID) + 1e-5f);
  o[tid]       = d0 * rstd * gam[tid] + bet[tid];
  o[tid + 256] = d1 * rstd * gam[tid + 256] + bet[tid + 256];
}

__global__ void build_tokens(const float* __restrict__ hfc, const float* __restrict__ cls,
                             float* __restrict__ tok, long long total) {
  long long g = blockIdx.x * 256LL + threadIdx.x;
  if (g >= total) return;
  const int c = (int)(g % HID);
  const long long bt = g / HID;
  const int t = (int)(bt % NTOK);
  const int b = (int)(bt / NTOK);
  float v;
  if (t == 0) v = cls[c];
  else { int s = t - 1; if (s >= 6000) s -= 6000; v = hfc[((long long)b * 6000 + s) * HID + c]; }
  tok[g] = v;
}

__global__ void split_qkv(const float* __restrict__ qkv, float* __restrict__ qh,
                          float* __restrict__ kh, float* __restrict__ vh, long long total) {
  long long g = blockIdx.x * 256LL + threadIdx.x;
  if (g >= total) return;
  const int c = (int)(g % HID);
  const long long bi = g / HID;          // b * NPAD + i
  const int i = (int)(bi % NPAD);
  const int b = (int)(bi / NPAD);
  const int h = c / DH, d = c % DH;
  const long long dst = (((long long)(b * HEADS + h)) * NPAD + i) * DH + d;
  const long long src = bi * (3 * HID) + c;
  qh[dst] = qkv[src] * 0.125f;           // DH^-0.5
  kh[dst] = qkv[src + HID];
  vh[dst] = qkv[src + 2 * HID];
}

__global__ void landmarks(const float* __restrict__ qh, const float* __restrict__ kh,
                          float* __restrict__ ql, float* __restrict__ kl, long long total) {
  long long g = blockIdx.x * 256LL + threadIdx.x;
  if (g >= total) return;
  const int d  = (int)(g % DH);
  const int m  = (int)((g / DH) % LMK);
  const int mh = (int)(g / ((long long)DH * LMK));     // 0..31
  const float* qb = qh + (((long long)mh * NPAD) + (long long)m * LSEG) * DH + d;
  const float* kb = kh + (((long long)mh * NPAD) + (long long)m * LSEG) * DH + d;
  float sq = 0.f, sk = 0.f;
  for (int j = 0; j < LSEG; ++j) { sq += qb[(long long)j * DH]; sk += kb[(long long)j * DH]; }
  ql[g] = sq * (1.f / LSEG);
  kl[g] = sk * (1.f / LSEG);
}

__global__ __launch_bounds__(256) void pinv_rowcol(const float* __restrict__ A, float* __restrict__ mx) {
  __shared__ float sr[256], sc[256];
  const float* a = A + (long long)blockIdx.x * (LMK * LMK);
  const int i = threadIdx.x;
  float rs = 0.f, cs = 0.f;
  for (int j = 0; j < LMK; ++j) { rs += fabsf(a[i * LMK + j]); cs += fabsf(a[j * LMK + i]); }
  sr[i] = rs; sc[i] = cs; __syncthreads();
  for (int s = 128; s > 0; s >>= 1) {
    if (i < s) { sr[i] = fmaxf(sr[i], sr[i + s]); sc[i] = fmaxf(sc[i], sc[i + s]); }
    __syncthreads();
  }
  if (i == 0) { mx[blockIdx.x] = sr[0]; mx[32 + blockIdx.x] = sc[0]; }
}

__global__ void pinv_combine(const float* __restrict__ mx, float* __restrict__ s) {
  if (threadIdx.x == 0) {
    float mr = 0.f, mc = 0.f;
    for (int i = 0; i < 32; ++i) { mr = fmaxf(mr, mx[i]); mc = fmaxf(mc, mx[32 + i]); }
    s[0] = 1.0f / (mr * mc);
  }
}

__global__ void zinit(const float* __restrict__ a2, const float* __restrict__ s,
                      float* __restrict__ z, long long total) {
  long long g = blockIdx.x * 256LL + threadIdx.x;
  if (g >= total) return;
  const int j = (int)(g % LMK);
  const int i = (int)((g / LMK) % LMK);
  const long long m = g / (LMK * LMK);
  z[g] = a2[(m * LMK + j) * LMK + i] * s[0];
}

__global__ void diag_sub(const float* __restrict__ in, float* __restrict__ out,
                         float c, long long total) {
  long long g = blockIdx.x * 256LL + threadIdx.x;
  if (g >= total) return;
  const int j = (int)(g % LMK);
  const int i = (int)((g / LMK) % LMK);
  out[g] = ((i == j) ? c : 0.f) - in[g];
}

__global__ void res_conv_add(float* __restrict__ outh, const float* __restrict__ vh,
                             const float* __restrict__ w, long long total) {
  long long g = blockIdx.x * 256LL + threadIdx.x;
  if (g >= total) return;
  const int d  = (int)(g % DH);
  const int i  = (int)((g / DH) % NPAD);
  const int mh = (int)(g / ((long long)DH * NPAD));
  const int h  = mh % HEADS;
  float acc = outh[g];
  const float* vb = vh + ((long long)mh * NPAD) * DH + d;
  for (int t = 0; t < 33; ++t) {
    const int ii = i + t - 16;
    if (ii >= 0 && ii < NPAD) acc += w[h * 33 + t] * vb[(long long)ii * DH];
  }
  outh[g] = acc;
}

__global__ void merge_heads(const float* __restrict__ outh, float* __restrict__ om, long long total) {
  long long g = blockIdx.x * 256LL + threadIdx.x;
  if (g >= total) return;
  const int c = (int)(g % HID);
  const long long bi = g / HID;
  const int i = (int)(bi % NPAD);
  const int b = (int)(bi / NPAD);
  const int h = c / DH, d = c % DH;
  om[g] = outh[(((long long)(b * HEADS + h)) * NPAD + i) * DH + d];
}

__global__ void residual_add(float* __restrict__ tok, const float* __restrict__ attn, long long total) {
  long long g = blockIdx.x * 256LL + threadIdx.x;
  if (g >= total) return;
  const int c = (int)(g % HID);
  const long long bt = g / HID;
  const int t = (int)(bt % NTOK);
  const int b = (int)(bt / NTOK);
  tok[g] += attn[(((long long)b * NPAD) + PADR + t) * HID + c];
}

__global__ void ppeg_cls(const float* __restrict__ tin, float* __restrict__ tout) {
  const int g = blockIdx.x * 256 + threadIdx.x;
  if (g >= BATCH * HID) return;
  const int c = g % HID, b = g / HID;
  tout[(long long)b * NTOK * HID + c] = tin[(long long)b * NTOK * HID + c];
}

__global__ void ppeg_conv(const float* __restrict__ tin, float* __restrict__ tout,
                          const float* __restrict__ w7, const float* __restrict__ b7,
                          const float* __restrict__ w5, const float* __restrict__ b5,
                          const float* __restrict__ w3, const float* __restrict__ b3,
                          long long total) {
  long long g = blockIdx.x * 256LL + threadIdx.x;
  if (g >= total) return;
  const int x = (int)(g % GRID_H);
  const int y = (int)((g / GRID_H) % GRID_H);
  const int c = (int)((g / (GRID_H * GRID_H)) % HID);
  const int b = (int)(g / ((long long)GRID_H * GRID_H * HID));
  const float* base = tin + ((long long)b * NTOK + 1) * HID + c;
  float acc = base[(long long)(y * GRID_H + x) * HID] + b7[c] + b5[c] + b3[c];
  const float* W7 = w7 + c * 49;
  const float* W5 = w5 + c * 25;
  const float* W3 = w3 + c * 9;
  for (int dy = -3; dy <= 3; ++dy) {
    const int yy = y + dy; if (yy < 0 || yy >= GRID_H) continue;
    for (int dx = -3; dx <= 3; ++dx) {
      const int xx = x + dx; if (xx < 0 || xx >= GRID_H) continue;
      const float f = base[(long long)(yy * GRID_H + xx) * HID];
      acc += W7[(dy + 3) * 7 + (dx + 3)] * f;
      if (dy >= -2 && dy <= 2 && dx >= -2 && dx <= 2) acc += W5[(dy + 2) * 5 + (dx + 2)] * f;
      if (dy >= -1 && dy <= 1 && dx >= -1 && dx <= 1) acc += W3[(dy + 1) * 3 + (dx + 1)] * f;
    }
  }
  tout[((long long)b * NTOK + 1 + y * GRID_H + x) * HID + c] = acc;
}

__global__ __launch_bounds__(256) void final_head(const float* __restrict__ tok,
                                                  const float* __restrict__ gam,
                                                  const float* __restrict__ bet,
                                                  const float* __restrict__ fc2_w,
                                                  const float* __restrict__ fc2_b,
                                                  float* __restrict__ xc,
                                                  float* __restrict__ out_mlp) {
  __shared__ float red[256];
  __shared__ float es[HID];
  const int b = blockIdx.x, tid = threadIdx.x;
  const float* x = tok + (long long)b * NTOK * HID;   // row 0 = cls token
  const float a0 = x[tid], a1 = x[tid + 256];
  red[tid] = a0 + a1; __syncthreads();
  for (int s = 128; s > 0; s >>= 1) { if (tid < s) red[tid] += red[tid + s]; __syncthreads(); }
  const float mu = red[0] * (1.f / HID); __syncthreads();
  const float d0 = a0 - mu, d1 = a1 - mu;
  red[tid] = d0 * d0 + d1 * d1; __syncthreads();
  for (int s = 128; s > 0; s >>= 1) { if (tid < s) red[tid] += red[tid + s]; __syncthreads(); }
  const float rstd = rsqrtf(red[0] * (1.f / HID) + 1e-5f);
  const float e0 = d0 * rstd * gam[tid] + bet[tid];
  const float e1 = d1 * rstd * gam[tid + 256] + bet[tid + 256];
  xc[(long long)b * HID + tid] = e0;
  xc[(long long)b * HID + tid + 256] = e1;
  es[tid] = e0; es[tid + 256] = e1; __syncthreads();
  if (tid < 2) {
    float s = fc2_b[tid];
    for (int k = 0; k < HID; ++k) s += es[k] * fc2_w[k * 2 + tid];
    out_mlp[b * 2 + tid] = s;
  }
}

__global__ void copy_rehearsal(const float* __restrict__ reh, float* __restrict__ xc, long long total) {
  long long g = blockIdx.x * 256LL + threadIdx.x;
  if (g >= total) return;
  xc[(long long)BATCH * HID + g] = reh[g];
}

__global__ void topk_softmax(const float* __restrict__ S, float* __restrict__ ew, int* __restrict__ idx) {
  const int row = blockIdx.x * 256 + threadIdx.x;
  if (row >= NNODE) return;
  const float* s = S + (long long)row * NNODE;
  float v[KNN]; int id[KNN];
#pragma unroll
  for (int k = 0; k < KNN; ++k) { v[k] = -3.4e38f; id[k] = 0; }
  for (int j = 0; j < NNODE; ++j) {
    const float xv = s[j];
    if (xv > v[KNN - 1]) {
      int p = KNN - 1;
      while (p > 0 && v[p - 1] < xv) { v[p] = v[p - 1]; id[p] = id[p - 1]; --p; }
      v[p] = xv; id[p] = j;
    }
  }
  float e[KNN]; float sum = 0.f;
#pragma unroll
  for (int k = 0; k < KNN; ++k) { e[k] = __expf(v[k] - v[0]); sum += e[k]; }
  const float inv = 1.f / sum;
#pragma unroll
  for (int k = 0; k < KNN; ++k) { ew[row * KNN + k] = e[k] * inv; idx[row * KNN + k] = id[k]; }
}

__global__ void gather_prop(const float* __restrict__ H, const float* __restrict__ ew,
                            const int* __restrict__ idx, float* __restrict__ out, long long total) {
  long long g = blockIdx.x * 256LL + threadIdx.x;
  if (g >= total) return;
  const int c   = (int)(g % HID);
  const int row = (int)(g / HID);
  float acc = 0.f;
#pragma unroll
  for (int k = 0; k < KNN; ++k)
    acc += ew[row * KNN + k] * H[(long long)idx[row * KNN + k] * HID + c];
  out[g] = acc;
}

__global__ void copy_graph_out(const float* __restrict__ gg, float* __restrict__ out) {
  const int t = threadIdx.x;
  if (t < 8) out[8 + t] = gg[t];   // first 4 rows x 2 logits
}

// ------------------------------ host side ----------------------------------
static inline unsigned blocks_for(long long tot) { return (unsigned)((tot + 255) / 256); }

static inline void gemm(hipStream_t st, const float* A, const float* B, const float* bias, float* C,
                        int M, int N, int K, int lda, int ldb, int ldc,
                        long long sA, long long sB, long long sC, int batch,
                        int transB, int act, float alpha) {
  dim3 g((unsigned)((N + BN - 1) / BN), (unsigned)((M + BM - 1) / BM), (unsigned)batch);
  wmma_gemm_f16<<<g, dim3(256), 0, st>>>(A, B, bias, C, M, N, K, lda, ldb, ldc,
                                         sA, sB, sC, transB, act, alpha);
}

static void run_nystrom(hipStream_t st,
                        const float* ln_g, const float* ln_b,
                        const float* qkv_w, const float* out_w, const float* out_b, const float* res_w,
                        float* tok,
                        float* lnp, float* qkvB, float* qh, float* kh, float* vh,
                        float* ql, float* kl, float* s1, float* a2, float* s3,
                        float* z, float* z2, float* az, float* t1, float* t2, float* ssc,
                        float* a3v, float* zv) {
  const int  MTOT = BATCH * NPAD;                  // 24576
  const long long sQK = (long long)NPAD * DH;      // per (b,h) q/k/v stride
  const long long sLM = (long long)LMK * DH;
  const long long sM  = (long long)LMK * LMK;

  // LN + front-pad to NPAD
  ln_pad<<<BATCH * NPAD, 256, 0, st>>>(tok, ln_g, ln_b, lnp);
  // qkv projection (no bias in reference)
  gemm(st, lnp, qkv_w, nullptr, qkvB, MTOT, 3 * HID, HID, HID, 3 * HID, 3 * HID,
       0, 0, 0, 1, 0, 0, 1.f);
  { long long tot = (long long)MTOT * HID;
    split_qkv<<<blocks_for(tot), 256, 0, st>>>(qkvB, qh, kh, vh, tot); }
  { long long tot = 32LL * LMK * DH;
    landmarks<<<blocks_for(tot), 256, 0, st>>>(qh, kh, ql, kl, tot); }

  // a1 = softmax(q @ kl^T)
  gemm(st, qh, kl, nullptr, s1, NPAD, LMK, DH, DH, DH, LMK,
       sQK, sLM, (long long)NPAD * LMK, 32, 1, 0, 1.f);
  row_softmax<<<32 * NPAD, 256, 0, st>>>(s1, LMK);
  // a2 = softmax(ql @ kl^T)
  gemm(st, ql, kl, nullptr, a2, LMK, LMK, DH, DH, DH, LMK,
       sLM, sLM, sM, 32, 1, 0, 1.f);
  row_softmax<<<32 * LMK, 256, 0, st>>>(a2, LMK);
  // a3 = softmax(ql @ k^T)
  gemm(st, ql, kh, nullptr, s3, LMK, NPAD, DH, DH, DH, NPAD,
       sLM, sQK, (long long)LMK * NPAD, 32, 1, 0, 1.f);
  row_softmax<<<32 * LMK, 256, 0, st>>>(s3, NPAD);

  // Moore-Penrose pinv of a2 (global scale across all b,h like the reference)
  pinv_rowcol<<<32, 256, 0, st>>>(a2, ssc);
  pinv_combine<<<1, 32, 0, st>>>(ssc, ssc + 64);
  { long long tot = 32LL * sM; zinit<<<blocks_for(tot), 256, 0, st>>>(a2, ssc + 64, z, tot); }
  float* zc = z; float* zn = z2;
  for (int it = 0; it < 6; ++it) {
    const long long tot = 32LL * sM;
    gemm(st, a2, zc, nullptr, az, LMK, LMK, LMK, LMK, LMK, LMK, sM, sM, sM, 32, 0, 0, 1.f);
    diag_sub<<<blocks_for(tot), 256, 0, st>>>(az, t1, 7.f, tot);
    gemm(st, az, t1, nullptr, t2, LMK, LMK, LMK, LMK, LMK, LMK, sM, sM, sM, 32, 0, 0, 1.f);
    diag_sub<<<blocks_for(tot), 256, 0, st>>>(t2, t1, 15.f, tot);
    gemm(st, az, t1, nullptr, t2, LMK, LMK, LMK, LMK, LMK, LMK, sM, sM, sM, 32, 0, 0, 1.f);
    diag_sub<<<blocks_for(tot), 256, 0, st>>>(t2, t1, 13.f, tot);
    gemm(st, zc, t1, nullptr, zn, LMK, LMK, LMK, LMK, LMK, LMK, sM, sM, sM, 32, 0, 0, 0.25f);
    float* tmp = zc; zc = zn; zn = tmp;
  }

  // a3v = a3 @ v ; zv = pinv @ a3v ; outh = a1 @ zv (reuse qh as outh)
  gemm(st, s3, vh, nullptr, a3v, LMK, DH, NPAD, NPAD, DH, DH,
       (long long)LMK * NPAD, sQK, sLM, 32, 0, 0, 1.f);
  gemm(st, zc, a3v, nullptr, zv, LMK, DH, LMK, LMK, DH, DH,
       sM, sLM, sLM, 32, 0, 0, 1.f);
  gemm(st, s1, zv, nullptr, qh, NPAD, DH, LMK, LMK, DH, DH,
       (long long)NPAD * LMK, sLM, sQK, 32, 0, 0, 1.f);

  // depthwise 33-tap residual conv on v, added into outh
  { long long tot = 32LL * NPAD * DH;
    res_conv_add<<<blocks_for(tot), 256, 0, st>>>(qh, vh, res_w, tot); }
  // merge heads (reuse lnp) and output projection (reuse qkvB)
  { long long tot = (long long)MTOT * HID;
    merge_heads<<<blocks_for(tot), 256, 0, st>>>(qh, lnp, tot); }
  gemm(st, lnp, out_w, out_b, qkvB, MTOT, HID, HID, HID, HID, HID, 0, 0, 0, 1, 0, 0, 1.f);
  // residual add of last NTOK rows into token buffer
  { long long tot = (long long)BATCH * NTOK * HID;
    residual_add<<<blocks_for(tot), 256, 0, st>>>(tok, qkvB, tot); }
}

extern "C" void kernel_launch(void* const* d_in, const int* in_sizes, int n_in,
                              void* d_out, int out_size, void* d_ws, size_t ws_size,
                              hipStream_t stream) {
  const float* x        = (const float*)d_in[0];
  const float* fc1_w    = (const float*)d_in[1];
  const float* fc1_b    = (const float*)d_in[2];
  const float* cls_tok  = (const float*)d_in[3];
  const float* l1_ng    = (const float*)d_in[4];
  const float* l1_nb    = (const float*)d_in[5];
  const float* l1_qkv_w = (const float*)d_in[6];
  const float* l1_out_w = (const float*)d_in[7];
  const float* l1_out_b = (const float*)d_in[8];
  const float* l1_res_w = (const float*)d_in[9];
  const float* l2_ng    = (const float*)d_in[10];
  const float* l2_nb    = (const float*)d_in[11];
  const float* l2_qkv_w = (const float*)d_in[12];
  const float* l2_out_w = (const float*)d_in[13];
  const float* l2_out_b = (const float*)d_in[14];
  const float* l2_res_w = (const float*)d_in[15];
  const float* pw7      = (const float*)d_in[16];
  const float* pb7      = (const float*)d_in[17];
  const float* pw5      = (const float*)d_in[18];
  const float* pb5      = (const float*)d_in[19];
  const float* pw3      = (const float*)d_in[20];
  const float* pb3      = (const float*)d_in[21];
  const float* nf_g     = (const float*)d_in[22];
  const float* nf_b     = (const float*)d_in[23];
  const float* fc2_w    = (const float*)d_in[24];
  const float* fc2_b    = (const float*)d_in[25];
  const float* rehearsal= (const float*)d_in[26];
  const float* agg_w    = (const float*)d_in[27];
  const float* agg_b    = (const float*)d_in[28];
  const float* g1_w     = (const float*)d_in[29];
  const float* g1_b     = (const float*)d_in[30];
  const float* g2_w     = (const float*)d_in[31];
  const float* g2_b     = (const float*)d_in[32];
  float* out = (float*)d_out;
  (void)in_sizes; (void)n_in; (void)out_size; (void)ws_size;

  // ---- workspace carve-out (all offsets 16B aligned) ----
  float* W = (float*)d_ws;
  size_t off = 0;
  auto alloc = [&](size_t n) -> float* { float* p = W + off; off += (n + 3) & ~(size_t)3; return p; };
  float* hfc  = alloc((size_t)BATCH * 6000 * HID);
  float* tokA = alloc((size_t)BATCH * NTOK * HID);
  float* tokB = alloc((size_t)BATCH * NTOK * HID);
  float* lnp  = alloc((size_t)BATCH * NPAD * HID);        // also merged-heads buffer
  float* qkvB = alloc((size_t)BATCH * NPAD * 3 * HID);    // also attention output
  float* qh   = alloc(32ULL * NPAD * DH);                 // also outh
  float* kh   = alloc(32ULL * NPAD * DH);
  float* vh   = alloc(32ULL * NPAD * DH);
  float* ql   = alloc(32ULL * LMK * DH);
  float* kl   = alloc(32ULL * LMK * DH);
  float* s1   = alloc(32ULL * NPAD * LMK);
  float* a2   = alloc(32ULL * LMK * LMK);
  float* s3   = alloc(32ULL * LMK * NPAD);
  float* z    = alloc(32ULL * LMK * LMK);
  float* z2   = alloc(32ULL * LMK * LMK);
  float* az   = alloc(32ULL * LMK * LMK);
  float* t1   = alloc(32ULL * LMK * LMK);
  float* t2   = alloc(32ULL * LMK * LMK);
  float* ssc  = alloc(128);
  float* a3v  = alloc(32ULL * LMK * DH);
  float* zv   = alloc(32ULL * LMK * DH);
  float* xc   = alloc((size_t)NNODE * HID);
  float* eb   = alloc((size_t)NNODE * HID);
  float* sc   = alloc((size_t)NNODE * NNODE);
  float* ew   = alloc((size_t)NNODE * KNN);
  int*   idxb = (int*)alloc((size_t)NNODE * KNN);
  float* p1   = alloc((size_t)NNODE * HID);
  float* h1   = alloc((size_t)NNODE * HID);
  float* p2   = alloc((size_t)NNODE * HID);
  float* gg   = alloc((size_t)NNODE * 2);

  // ---- fc1 + relu, assemble [cls | feat | wrap] tokens ----
  gemm(stream, x, fc1_w, fc1_b, hfc, BATCH * 6000, HID, 1024, 1024, HID, HID,
       0, 0, 0, 1, 0, 1, 1.f);
  { long long tot = (long long)BATCH * NTOK * HID;
    build_tokens<<<blocks_for(tot), 256, 0, stream>>>(hfc, cls_tok, tokA, tot); }

  // ---- layer 1: Nystrom attention ----
  run_nystrom(stream, l1_ng, l1_nb, l1_qkv_w, l1_out_w, l1_out_b, l1_res_w,
              tokA, lnp, qkvB, qh, kh, vh, ql, kl, s1, a2, s3,
              z, z2, az, t1, t2, ssc, a3v, zv);

  // ---- PPEG (fused 7x7 + 5x5 + 3x3 depthwise + identity) ----
  { long long tot = (long long)BATCH * HID * GRID_H * GRID_H;
    ppeg_conv<<<blocks_for(tot), 256, 0, stream>>>(tokA, tokB, pw7, pb7, pw5, pb5, pw3, pb3, tot); }
  ppeg_cls<<<blocks_for((long long)BATCH * HID), 256, 0, stream>>>(tokA, tokB);

  // ---- layer 2: Nystrom attention ----
  run_nystrom(stream, l2_ng, l2_nb, l2_qkv_w, l2_out_w, l2_out_b, l2_res_w,
              tokB, lnp, qkvB, qh, kh, vh, ql, kl, s1, a2, s3,
              z, z2, az, t1, t2, ssc, a3v, zv);

  // ---- final LN on cls + MLP logits; emb placed as xc rows 0..3 ----
  final_head<<<BATCH, 256, 0, stream>>>(tokB, nf_g, nf_b, fc2_w, fc2_b, xc, out);
  { long long tot = 1024LL * HID;
    copy_rehearsal<<<blocks_for(tot), 256, 0, stream>>>(rehearsal, xc, tot); }

  // ---- graph head ----
  gemm(stream, xc, agg_w, agg_b, eb, NNODE, HID, HID, HID, HID, HID, 0, 0, 0, 1, 0, 2, 1.f);
  gemm(stream, eb, eb, nullptr, sc, NNODE, NNODE, HID, HID, HID, NNODE, 0, 0, 0, 1, 1, 0, 1.f);
  topk_softmax<<<blocks_for(NNODE), 256, 0, stream>>>(sc, ew, idxb);
  { long long tot = (long long)NNODE * HID;
    gather_prop<<<blocks_for(tot), 256, 0, stream>>>(xc, ew, idxb, p1, tot); }
  gemm(stream, p1, g1_w, g1_b, h1, NNODE, HID, HID, HID, HID, HID, 0, 0, 0, 1, 0, 1, 1.f);
  { long long tot = (long long)NNODE * HID;
    gather_prop<<<blocks_for(tot), 256, 0, stream>>>(h1, ew, idxb, p2, tot); }
  gemm(stream, p2, g2_w, g2_b, gg, NNODE, 2, HID, HID, 2, 2, 0, 0, 0, 1, 0, 0, 1.f);
  copy_graph_out<<<1, 32, 0, stream>>>(gg, out);
}
